// FullyConnectedPINN_23845658427755
// MI455X (gfx1250) — matrix operand order repaired and checked by reading, not compiled, not run on hardware
//
#include <hip/hip_runtime.h>
#include <stdint.h>

typedef __attribute__((ext_vector_type(16))) __bf16 v16bf;
typedef __attribute__((ext_vector_type(8)))  float  v8f;
typedef __attribute__((ext_vector_type(4)))  unsigned int v4u;
typedef __attribute__((ext_vector_type(8)))  int v8i;
typedef __attribute__((ext_vector_type(4)))  int v4i;

#define B_TOTAL     262144
#define ROWS_WG     256                 // 8 waves * 32 rows
#define W0_ELEMS    (512*256)
#define WH_ELEMS    (256*256)
#define CHUNK_BYTES 65536               // one TDM chunk = 4 ksteps * 16 ntiles * 1KB
#define CHUNK_U16   (CHUNK_BYTES/2)
#define NCHUNK      18                  // L0: 4 chunks, 7 hidden layers * 2 chunks

// ---------- helpers ----------
__device__ __forceinline__ uint16_t f2bf(float f) {
  uint32_t u = __float_as_uint(f);
  uint32_t r = (u + 0x7FFFu + ((u >> 16) & 1u)) >> 16;  // RNE truncate
  return (uint16_t)r;
}
__device__ __forceinline__ float bf2f(uint16_t h) {
  return __uint_as_float(((uint32_t)h) << 16);
}
__device__ __forceinline__ float fast_tanh(float x) {
  x = fminf(10.0f, fmaxf(-10.0f, x));
  float e = __builtin_amdgcn_exp2f(x * 2.8853900817779268f);  // e^{2x}
  return (e - 1.0f) * __builtin_amdgcn_rcpf(e + 1.0f);
}

struct AFrag { uint4 lo, hi; };   // 32 bytes == v16bf

// ---------- prep: repack fp32 weights into bf16 WMMA B-fragments ----------
// uint16 index = layerBase + ((kstep*16 + ntile)*32 + lane)*16 + j
// lane l -> column n = ntile*16 + (l&15), k = kstep*32 + ((l>>4)<<4) + j
__global__ void pack_weights(const float* __restrict__ W0,
                             const float* __restrict__ Wh,
                             uint16_t* __restrict__ ws) {
  int gid = blockIdx.x * blockDim.x + threadIdx.x;
  if (gid >= W0_ELEMS + 7 * WH_ELEMS) return;
  const float* src; int flat, dstbase;
  if (gid < W0_ELEMS) { src = W0; flat = gid; dstbase = 0; }
  else {
    int rem   = gid - W0_ELEMS;
    int layer = rem / WH_ELEMS;
    flat      = rem % WH_ELEMS;
    src       = Wh + layer * WH_ELEMS;
    dstbase   = W0_ELEMS + layer * WH_ELEMS;
  }
  int j     = flat & 15;
  int lane  = (flat >> 4) & 31;
  int ntile = (flat >> 9) & 15;
  int kstep = flat >> 13;
  int n = ntile * 16 + (lane & 15);
  int k = kstep * 32 + ((lane >> 4) << 4) + j;
  ws[dstbase + flat] = f2bf(src[k * 256 + n]);
}

// ---------- TDM: async global->LDS weight staging (ISA ch.8 D# layout) ----------
__device__ __forceinline__ void tdm_issue_chunk(const uint16_t* __restrict__ wfrag,
                                                uint16_t* sW, int c) {
#if __has_builtin(__builtin_amdgcn_tensor_load_to_lds)
  uint64_t ga  = (uint64_t)(uintptr_t)(wfrag + (size_t)c * CHUNK_U16);
  uint32_t lds = (uint32_t)(uintptr_t)(sW + (c & 1) * CHUNK_U16);  // flat[31:0] == LDS offset
  const uint32_t n8 = CHUNK_BYTES >> 3;                            // 8192 elements of 8B
  v4u g0;
  g0.x = 1u;                                                 // count=1, user mode
  g0.y = lds;                                                // lds_addr
  g0.z = (uint32_t)ga;                                       // global_addr[31:0]
  g0.w = (uint32_t)((ga >> 32) & 0x1FFFFFFu) | (2u << 30);   // global_addr[56:32] | type=2
  v8i g1;
  g1[0] = (int)(3u << 16);          // data_size = 8B, no multicast/pad/iterate
  g1[1] = (int)(n8 << 16);          // tensor_dim0[15:0] at bits [63:48]
  g1[2] = (int)(1u << 16);          // tensor_dim0[31:16]=0, tensor_dim1=1
  g1[3] = (int)(n8 << 16);          // tile_dim0 at bits [127:112]
  g1[4] = 1;                        // tile_dim1 = 1, tile_dim2 = 0
  g1[5] = (int)n8;                  // tensor_dim0_stride[31:0]
  g1[6] = 0;
  g1[7] = 0;
  v4i z = {0, 0, 0, 0};
#if __clang_major__ >= 23
  v8i z8 = {0, 0, 0, 0, 0, 0, 0, 0};
  __builtin_amdgcn_tensor_load_to_lds(g0, g1, z, z, z8, 0);
#else
  __builtin_amdgcn_tensor_load_to_lds(g0, g1, z, z, 0);
#endif
#else
  (void)wfrag; (void)sW; (void)c;
#endif
}

__device__ __forceinline__ void tdm_complete_chunk(const uint16_t* __restrict__ wfrag,
                                                   uint16_t* sW, int c, int wave) {
#if __has_builtin(__builtin_amdgcn_tensor_load_to_lds)
  if (wave == 0) __builtin_amdgcn_s_wait_tensorcnt(0);
  (void)wfrag; (void)sW; (void)c;
#else
  if (c < NCHUNK) {   // cooperative fallback: all 256 threads copy the chunk
    const uint4* gs = (const uint4*)(wfrag + (size_t)c * CHUNK_U16);
    uint4*       ls = (uint4*)(sW + (c & 1) * CHUNK_U16);
    for (int i = threadIdx.x; i < CHUNK_BYTES / 16; i += 256) ls[i] = gs[i];
  }
  (void)wave;
#endif
}

// ---------- WMMA micro-steps ----------
__device__ __forceinline__ v16bf load_afrag(const uint16_t* __restrict__ actb,
                                            int kcol, int lane) {
  // A 16x32 in this file's contiguous k order (same as packed B): lane<16 -> row=lane, K[0..15]; lane>=16 -> K[16..31]
  const uint16_t* ap = actb + (lane & 15) * 256 + kcol + (lane >> 4) * 16;
  AFrag af;
  af.lo = *(const uint4*)(ap);
  af.hi = *(const uint4*)(ap + 8);
  return __builtin_bit_cast(v16bf, af);
}

// one k-step for two M-tiles, two n-tiles per iteration with two live B buffers:
// 4 DS loads issue together, so the first WMMA pair only needs dscnt<=2.
__device__ __forceinline__ void wmma_step2(v8f acc0[16], v8f acc1[16],
                                           v16bf a0, v16bf a1,
                                           const uint16_t* __restrict__ wst,
                                           int k4, int lane) {
  const uint16_t* wp = wst + (k4 * 16) * 512 + lane * 16;
#pragma unroll
  for (int n = 0; n < 16; n += 2) {
    v16bf b0 = *(const v16bf*)(wp + n * 512);
    v16bf b1 = *(const v16bf*)(wp + (n + 1) * 512);
    acc0[n] = __builtin_amdgcn_wmma_f32_16x16x32_bf16(
        false, a0, false, b0, (short)0, acc0[n], false, false);
    acc1[n] = __builtin_amdgcn_wmma_f32_16x16x32_bf16(
        false, a1, false, b0, (short)0, acc1[n], false, false);
    acc0[n + 1] = __builtin_amdgcn_wmma_f32_16x16x32_bf16(
        false, a0, false, b1, (short)0, acc0[n + 1], false, false);
    acc1[n + 1] = __builtin_amdgcn_wmma_f32_16x16x32_bf16(
        false, a1, false, b1, (short)0, acc1[n + 1], false, false);
  }
}

// accumulate one 64KB weight chunk (4 ksteps) for this wave's 32 rows
__device__ __forceinline__ void chunk_accum(v8f acc0[16], v8f acc1[16],
                                            const uint16_t* __restrict__ slab,
                                            int colbase,
                                            const uint16_t* __restrict__ sW,
                                            int c, int lane) {
  const uint16_t* wst = sW + (c & 1) * CHUNK_U16;
#pragma unroll
  for (int k4 = 0; k4 < 4; ++k4) {
    int kcol = colbase + k4 * 32;
    v16bf a0 = load_afrag(slab, kcol, lane);             // rows 0..15
    v16bf a1 = load_afrag(slab + 16 * 256, kcol, lane);  // rows 16..31
    wmma_step2(acc0, acc1, a0, a1, wst, k4, lane);
  }
}

__device__ __forceinline__ void epilogue_tanh2(v8f acc0[16], v8f acc1[16],
                                               const float* __restrict__ bias,
                                               uint16_t* __restrict__ slab, int lane) {
#pragma unroll
  for (int n = 0; n < 16; ++n) {
    int   col   = n * 16 + (lane & 15);
    float bs    = bias[col];
    int   rbase = (lane >> 4) * 8;                 // lanes 16-31 hold M=8..15
#pragma unroll
    for (int g = 0; g < 8; ++g) {
      slab[(rbase + g) * 256 + col]        = f2bf(fast_tanh(acc0[n][g] + bs));
      slab[(rbase + g + 16) * 256 + col]   = f2bf(fast_tanh(acc1[n][g] + bs));
    }
  }
}

// ---------- fused PINN forward ----------
__global__ __launch_bounds__(256, 1)
void pinn_fused(const float* __restrict__ x,
                const float* __restrict__ proj,
                const float* __restrict__ b0,
                const float* __restrict__ bh,
                const float* __restrict__ Wout,
                const float* __restrict__ bout,
                const uint16_t* __restrict__ wfrag,
                float* __restrict__ out) {
  __shared__ __align__(16) uint16_t sAct[8 * 32 * 256];    // 128 KB, in-place activations
  __shared__ __align__(16) uint16_t sW[2 * CHUNK_U16];     // 128 KB, TDM double buffer

  const int wave    = threadIdx.x >> 5;
  const int lane    = threadIdx.x & 31;
  const int rowbase = blockIdx.x * ROWS_WG + wave * 32;
  uint16_t* slab = sAct + wave * (32 * 256);

  // kick off chunk 0 staging, overlapped with the sin encode
  if (wave == 0) tdm_issue_chunk(wfrag, sW, 0);

  // ---- encode pass (sin): v_sin takes revolutions -> feed x@proj raw
  for (int t = lane; t < 32 * 256; t += 32) {
    int r = t >> 8;
    int j = t & 255;
    float p = x[(rowbase + r) * 2 + 0] * proj[j] +
              x[(rowbase + r) * 2 + 1] * proj[256 + j];
    slab[r * 256 + j] = f2bf(__builtin_amdgcn_sinf(p));
  }
  tdm_complete_chunk(wfrag, sW, 0, wave);
  __syncthreads();

  int c = 0;
  v8f acc0[16], acc1[16];
#pragma unroll
  for (int n = 0; n < 16; ++n) {
    acc0[n] = v8f{0.f,0.f,0.f,0.f,0.f,0.f,0.f,0.f};
    acc1[n] = v8f{0.f,0.f,0.f,0.f,0.f,0.f,0.f,0.f};
  }

  // ---- layer 0, sin half: chunks 0..1 cover W0 K=0..255 ----
#pragma unroll 1
  for (int kc = 0; kc < 2; ++kc) {
    if (wave == 0 && c + 1 < NCHUNK) tdm_issue_chunk(wfrag, sW, c + 1);
    chunk_accum(acc0, acc1, slab, kc * 128, sW, c, lane);
    __syncthreads();
    tdm_complete_chunk(wfrag, sW, c + 1, wave);
    __syncthreads();
    ++c;
  }

  // ---- re-encode same buffer with cos (all sin reads are done; wave-private) ----
  for (int t = lane; t < 32 * 256; t += 32) {
    int r = t >> 8;
    int j = t & 255;
    float p = x[(rowbase + r) * 2 + 0] * proj[j] +
              x[(rowbase + r) * 2 + 1] * proj[256 + j];
    slab[r * 256 + j] = f2bf(__builtin_amdgcn_cosf(p));
  }

  // ---- layer 0, cos half: chunks 2..3 cover W0 K=256..511 ----
#pragma unroll 1
  for (int kc = 0; kc < 2; ++kc) {
    if (wave == 0 && c + 1 < NCHUNK) tdm_issue_chunk(wfrag, sW, c + 1);
    chunk_accum(acc0, acc1, slab, kc * 128, sW, c, lane);
    __syncthreads();
    tdm_complete_chunk(wfrag, sW, c + 1, wave);
    __syncthreads();
    ++c;
  }
  epilogue_tanh2(acc0, acc1, b0, slab, lane);   // in-place: all reads precede writes

  // ---- hidden layers 1..7: in-place, 2 chunks each ----
#pragma unroll 1
  for (int l = 0; l < 7; ++l) {
#pragma unroll
    for (int n = 0; n < 16; ++n) {
      acc0[n] = v8f{0.f,0.f,0.f,0.f,0.f,0.f,0.f,0.f};
      acc1[n] = v8f{0.f,0.f,0.f,0.f,0.f,0.f,0.f,0.f};
    }
#pragma unroll 1
    for (int kc = 0; kc < 2; ++kc) {
      if (wave == 0 && c + 1 < NCHUNK) tdm_issue_chunk(wfrag, sW, c + 1);
      chunk_accum(acc0, acc1, slab, kc * 128, sW, c, lane);
      __syncthreads();
      tdm_complete_chunk(wfrag, sW, c + 1, wave);
      __syncthreads();
      ++c;
    }
    epilogue_tanh2(acc0, acc1, bh + l * 256, slab, lane);
  }

  // ---- output layer: [32,256] x [256,1] via VALU + cross-half shuffle ----
#pragma unroll 1
  for (int m = 0; m < 2; ++m) {
    int   r    = m * 16 + (lane & 15);
    int   half = lane >> 4;
    float sum  = 0.f;
    const uint16_t* hr = slab + r * 256 + half * 128;
    const float*    wo = Wout + half * 128;
    for (int k = 0; k < 128; ++k) sum += bf2f(hr[k]) * wo[k];
    sum += __shfl_xor(sum, 16);
    if (lane < 16) out[rowbase + r] = sum + bout[0];
  }
}

extern "C" void kernel_launch(void* const* d_in, const int* in_sizes, int n_in,
                              void* d_out, int out_size, void* d_ws, size_t ws_size,
                              hipStream_t stream) {
  const float* x    = (const float*)d_in[0];
  const float* proj = (const float*)d_in[1];
  const float* W0   = (const float*)d_in[2];
  const float* b0   = (const float*)d_in[3];
  const float* Wh   = (const float*)d_in[4];
  const float* bh   = (const float*)d_in[5];
  const float* Wout = (const float*)d_in[6];
  const float* bout = (const float*)d_in[7];
  uint16_t* ws  = (uint16_t*)d_ws;
  float*    out = (float*)d_out;

  int packN = W0_ELEMS + 7 * WH_ELEMS;
  pack_weights<<<(packN + 255) / 256, 256, 0, stream>>>(W0, Wh, ws);
  pinn_fused<<<B_TOTAL / ROWS_WG, 256, 0, stream>>>(x, proj, b0, bh, Wout, bout, ws, out);
}
